// GATv4_26414048870728
// MI455X (gfx1250) — matrix operand order repaired and checked by reading, not compiled, not run on hardware
//
#include <hip/hip_runtime.h>
#include <hip/hip_bf16.h>

// ---------------- problem constants (from reference) ----------------
#define Bv     16
#define Nv     2048
#define IN_CH  128
#define Hh     4
#define Cc     64
#define HC     256
#define Ev     524288
#define NT     32768          // B*N
#define EP     (Ev + NT)      // edges + self loops = 557056
#define FC0    1024
#define FC1    512
#define OMIC   128
#define OUTN   2
#define LN_EPS 1e-5f

typedef __attribute__((ext_vector_type(16))) _Float16 v16h;
typedef __attribute__((ext_vector_type(8)))  float    v8f;

// ---------------- small helpers ----------------
__device__ __forceinline__ void atomicMaxF(float* addr, float val) {
    unsigned int* ua = (unsigned int*)addr;
    unsigned int old = *ua;
    while (true) {
        float f = __uint_as_float(old);
        if (f >= val) break;
        unsigned int assumed = old;
        old = atomicCAS(ua, assumed, __float_as_uint(val));
        if (old == assumed) break;
    }
}

__device__ __forceinline__ float elu1(float x) {
    return x > 0.f ? x : (__expf(x) - 1.f);
}

// ---------------- 1) per-node mean of x over 128 channels ----------------
__global__ void mean_rows_kernel(const float* __restrict__ x, float* __restrict__ out) {
    int tid = blockIdx.x * blockDim.x + threadIdx.x;
    if (tid >= NT) return;
    const float4* xp = (const float4*)(x + (size_t)tid * IN_CH);
    float s = 0.f;
#pragma unroll 8
    for (int i = 0; i < IN_CH / 4; ++i) {
        float4 v = xp[i];
        s += v.x + v.y + v.z + v.w;
    }
    out[tid] = s * (1.f / (float)IN_CH);
}

// ---------------- 2) LayerNorm over rows of length Nv, write into feat slice --
__global__ void ln_rows_kernel(const float* __restrict__ in, const float* __restrict__ w,
                               const float* __restrict__ b, float* __restrict__ out,
                               int outStride, int outOff) {
    __shared__ float red[256];
    const int row = blockIdx.x;
    const float* p = in + (size_t)row * Nv;
    float s = 0.f;
    for (int i = threadIdx.x; i < Nv; i += 256) s += p[i];
    red[threadIdx.x] = s; __syncthreads();
    for (int o = 128; o > 0; o >>= 1) {
        if (threadIdx.x < o) red[threadIdx.x] += red[threadIdx.x + o];
        __syncthreads();
    }
    float mu = red[0] * (1.f / (float)Nv);
    __syncthreads();
    float v = 0.f;
    for (int i = threadIdx.x; i < Nv; i += 256) { float d = p[i] - mu; v += d * d; }
    red[threadIdx.x] = v; __syncthreads();
    for (int o = 128; o > 0; o >>= 1) {
        if (threadIdx.x < o) red[threadIdx.x] += red[threadIdx.x + o];
        __syncthreads();
    }
    float rstd = rsqrtf(red[0] * (1.f / (float)Nv) + LN_EPS);
    __syncthreads();
    for (int i = threadIdx.x; i < Nv; i += 256)
        out[(size_t)row * outStride + outOff + i] = (p[i] - mu) * rstd * w[i] + b[i];
}

// ---------------- 3) WMMA GEMM: C[M,N] = A[M,K] @ W[K,N] (f32 in/out, f16 MAC)
// K compile-time (128/256) so the K-loop fully unrolls.
// Block = 256 threads = 8 wave32s; block computes a 128(M) x 64(N) C tile.
// Each wave computes 16x64 with 4 accumulators: one A fragment feeds 4 WMMAs.
// B is staged in LDS pre-packed in WMMA fragment layout: each lane's fragment
// is 32 contiguous bytes -> two ds_load_b128 per fragment.
// grid.x = N/64, grid.y = M/128.
template <int K>
__global__ void __launch_bounds__(256) wmma_gemm_kernel(const float* __restrict__ A,
                                                        const float* __restrict__ W,
                                                        float* __restrict__ C,
                                                        int N) {
    constexpr int KSTEPS = K / 32;
    // [kstep][ntile][lane][16 halves] ; K*64 halves (<=32KB)
    __shared__ __align__(32) _Float16 sB[KSTEPS * 4 * 32 * 16];

    // ---- cooperative stage: scatter W into fragment layout, f32 -> f16 ----
    // t = ((kst*4 + nt)*32 + lane)*16 + i
    // element (k, n): k = kst*32 + (i>>3)*16 + (lane>>4)*8 + (i&7) ; n = nt*16 + (lane&15)
    const int nBase = blockIdx.x * 64;
    for (int t = threadIdx.x; t < KSTEPS * 4 * 32 * 16; t += 256) {
        int i  = t & 15;
        int ln = (t >> 4) & 31;
        int nt = (t >> 9) & 3;
        int ks = t >> 11;
        int k  = ks * 32 + (i >> 3) * 16 + (ln >> 4) * 8 + (i & 7);
        int n  = nt * 16 + (ln & 15);
        sB[t] = (_Float16)W[(size_t)k * N + nBase + n];
    }
    __syncthreads();

    const int wave = threadIdx.x >> 5;
    const int lane = threadIdx.x & 31;
    const int tm   = blockIdx.y * 8 + wave;
    const int row  = lane & 15;      // M row for A frag / N col within tile for C
    const int grp  = lane >> 4;      // lane half-group per ISA layout

    const float* arow = A + (size_t)(tm * 16 + row) * K;
    v8f acc0 = {}, acc1 = {}, acc2 = {}, acc3 = {};
#pragma unroll
    for (int ks = 0; ks < KSTEPS; ++ks) {
        // A fragment per ISA 16-bit 16x32 layout (two 8-float contiguous runs)
        v16h a;
        const float* ap = arow + ks * 32 + grp * 8;
#pragma unroll
        for (int i = 0; i < 8; ++i) {
            a[i]     = (_Float16)ap[i];        // K = ks*32 + grp*8 + i
            a[8 + i] = (_Float16)ap[16 + i];   // K = ks*32 + 16 + grp*8 + i
        }
        const v16h b0 = *(const v16h*)(sB + (((ks * 4 + 0) * 32 + lane) << 4));
        const v16h b1 = *(const v16h*)(sB + (((ks * 4 + 1) * 32 + lane) << 4));
        const v16h b2 = *(const v16h*)(sB + (((ks * 4 + 2) * 32 + lane) << 4));
        const v16h b3 = *(const v16h*)(sB + (((ks * 4 + 3) * 32 + lane) << 4));
        acc0 = __builtin_amdgcn_wmma_f32_16x16x32_f16(false, a, false, b0, (short)0, acc0, false, false);
        acc1 = __builtin_amdgcn_wmma_f32_16x16x32_f16(false, a, false, b1, (short)0, acc1, false, false);
        acc2 = __builtin_amdgcn_wmma_f32_16x16x32_f16(false, a, false, b2, (short)0, acc2, false, false);
        acc3 = __builtin_amdgcn_wmma_f32_16x16x32_f16(false, a, false, b3, (short)0, acc3, false, false);
    }
    // C/D layout: VGPR r -> (M = grp*8 + r, N = lane&15)
    float* cbase = C + (size_t)(tm * 16 + grp * 8) * N + nBase + row;
#pragma unroll
    for (int r = 0; r < 8; ++r) {
        cbase[(size_t)r * N +  0] = acc0[r];
        cbase[(size_t)r * N + 16] = acc1[r];
        cbase[(size_t)r * N + 32] = acc2[r];
        cbase[(size_t)r * N + 48] = acc3[r];
    }
}

// ---------------- 4) attention scores a_src/a_dst : [NT,H] ----------------
__global__ void attn_scores_kernel(const float* __restrict__ xw,
                                   const float* __restrict__ att_src,
                                   const float* __restrict__ att_dst,
                                   float* __restrict__ asrc, float* __restrict__ adst) {
    int tid = blockIdx.x * blockDim.x + threadIdx.x;
    if (tid >= NT * Hh) return;
    int n = tid >> 2, h = tid & 3;
    const float* xp = xw + (size_t)n * HC + h * Cc;
    const float* as = att_src + h * Cc;
    const float* ad = att_dst + h * Cc;
    float s1 = 0.f, s2 = 0.f;
#pragma unroll 8
    for (int c = 0; c < Cc; ++c) { s1 += xp[c] * as[c]; s2 += xp[c] * ad[c]; }
    asrc[tid] = s1; adst[tid] = s2;
}

__global__ void fill_neginf_kernel(float* __restrict__ p, int n) {
    int tid = blockIdx.x * blockDim.x + threadIdx.x;
    if (tid < n) p[tid] = -3.0e38f;
}

// ---------------- 5) edge passes ----------------
__device__ __forceinline__ void edge_sd(const long long* ei, int e, int& s, int& d) {
    if (e < Ev) { s = (int)ei[e]; d = (int)ei[Ev + e]; }
    else        { s = e - Ev; d = e - Ev; }   // self loops
}

__global__ void edge_alpha_max_kernel(const long long* __restrict__ ei,
                                      const float* __restrict__ asrc,
                                      const float* __restrict__ adst,
                                      float* __restrict__ alpha, float* __restrict__ amax) {
    int e = blockIdx.x * blockDim.x + threadIdx.x;
    if (e >= EP) return;
    int s, d; edge_sd(ei, e, s, d);
#pragma unroll
    for (int h = 0; h < Hh; ++h) {
        float v = asrc[s * Hh + h] + adst[d * Hh + h];
        v = v > 0.f ? v : 0.2f * v;          // leaky_relu(0.2)
        alpha[(size_t)e * Hh + h] = v;
        atomicMaxF(&amax[d * Hh + h], v);
    }
}

__global__ void edge_exp_sum_kernel(const long long* __restrict__ ei,
                                    float* __restrict__ alpha,
                                    const float* __restrict__ amax,
                                    float* __restrict__ denom) {
    int e = blockIdx.x * blockDim.x + threadIdx.x;
    if (e >= EP) return;
    int s, d; edge_sd(ei, e, s, d); (void)s;
#pragma unroll
    for (int h = 0; h < Hh; ++h) {
        float ea = __expf(alpha[(size_t)e * Hh + h] - amax[d * Hh + h]);
        alpha[(size_t)e * Hh + h] = ea;      // reuse as exp(alpha)
        atomicAdd(&denom[d * Hh + h], ea);
    }
}

// one wave32 per edge: lane covers 8 contiguous channels (single head per lane)
__global__ void edge_message_kernel(const long long* __restrict__ ei,
                                    const float* __restrict__ xw,
                                    const float* __restrict__ ea,
                                    const float* __restrict__ denom,
                                    float* __restrict__ accum) {
    int gid  = blockIdx.x * blockDim.x + threadIdx.x;
    int e    = gid >> 5;
    int lane = gid & 31;
    if (e >= EP) return;
    int s, d; edge_sd(ei, e, s, d);
    int h = lane >> 3;                       // 8 lanes per head (64 ch / 8)
    float coef = ea[(size_t)e * Hh + h] / (denom[d * Hh + h] + 1e-16f);
    const float4* sp = (const float4*)(xw + (size_t)s * HC) + lane * 2;
    float4 v0 = sp[0], v1 = sp[1];
    float* dp = accum + (size_t)d * HC + lane * 8;
    atomicAdd(dp + 0, v0.x * coef); atomicAdd(dp + 1, v0.y * coef);
    atomicAdd(dp + 2, v0.z * coef); atomicAdd(dp + 3, v0.w * coef);
    atomicAdd(dp + 4, v1.x * coef); atomicAdd(dp + 5, v1.y * coef);
    atomicAdd(dp + 6, v1.z * coef); atomicAdd(dp + 7, v1.w * coef);
}

// ---------------- 6) bias + ELU in place, fused pool projection ----------------
__global__ void elu_pool_kernel(float* __restrict__ h, const float* __restrict__ bias,
                                const float* __restrict__ pw, const float* __restrict__ pb,
                                float* __restrict__ pre) {
    int n = blockIdx.x * blockDim.x + threadIdx.x;
    if (n >= NT) return;
    float* hp = h + (size_t)n * HC;
    float acc = 0.f;
#pragma unroll 4
    for (int j = 0; j < HC; ++j) {
        float v = elu1(hp[j] + bias[j]);
        hp[j] = v;
        acc += v * pw[j];
    }
    pre[n] = acc + pb[0];
}

// ---------------- 7) plain FC (M=16 rows; WMMA not worth it here) -------------
__global__ void fc_kernel(const float* __restrict__ in, const float* __restrict__ W,
                          const float* __restrict__ bias, float* __restrict__ out,
                          int rows, int K, int N, int doElu) {
    int tid = blockIdx.x * blockDim.x + threadIdx.x;
    if (tid >= rows * N) return;
    int r = tid / N, n = tid % N;
    const float* ip = in + (size_t)r * K;
    float s = bias[n];
    for (int k = 0; k < K; ++k) s += ip[k] * W[(size_t)k * N + n];
    if (doElu) s = elu1(s);
    out[tid] = s;
}

// ---------------- launch ----------------
extern "C" void kernel_launch(void* const* d_in, const int* in_sizes, int n_in,
                              void* d_out, int out_size, void* d_ws, size_t ws_size,
                              hipStream_t stream) {
    const float*     x        = (const float*)d_in[0];
    const long long* ei       = (const long long*)d_in[1];
    const float*     w1       = (const float*)d_in[2];
    const float*     att_src1 = (const float*)d_in[3];
    const float*     att_dst1 = (const float*)d_in[4];
    const float*     bias1    = (const float*)d_in[5];
    const float*     pool1_w  = (const float*)d_in[6];
    const float*     pool1_b  = (const float*)d_in[7];
    const float*     w2       = (const float*)d_in[8];
    const float*     att_src2 = (const float*)d_in[9];
    const float*     att_dst2 = (const float*)d_in[10];
    const float*     bias2    = (const float*)d_in[11];
    const float*     pool2_w  = (const float*)d_in[12];
    const float*     pool2_b  = (const float*)d_in[13];
    const float*     ln_w     = (const float*)d_in[14];
    const float*     ln_b     = (const float*)d_in[15];
    const float*     enc0_w   = (const float*)d_in[16];
    const float*     enc0_b   = (const float*)d_in[17];
    const float*     enc1_w   = (const float*)d_in[18];
    const float*     enc1_b   = (const float*)d_in[19];
    const float*     enc2_w   = (const float*)d_in[20];
    const float*     enc2_b   = (const float*)d_in[21];
    const float*     last_w   = (const float*)d_in[22];
    const float*     last_b   = (const float*)d_in[23];
    float*           out      = (float*)d_out;

    // ---- workspace layout (bytes) ----
    char* base = (char*)d_ws;
    float* bufA  = (float*)(base + 0);                 // NT*HC (xw layer1, then xw layer2)
    float* bufB  = (float*)(base + 33554432);          // NT*HC (accum/h layer1, then layer2)
    float* alpha = (float*)(base + 67108864);          // EP*H
    float* asrc  = (float*)(base + 76021760);          // NT*H
    float* adst  = (float*)(base + 76546048);          // NT*H
    float* amax  = (float*)(base + 77070336);          // NT*H
    float* denom = (float*)(base + 77594624);          // NT*H
    float* nmean = (float*)(base + 78118912);          // NT (also pre-LN temp)
    float* feat  = (float*)(base + 78249984);          // B*3N
    float* e0    = (float*)(base + 78643200);          // B*FC0
    float* e1    = (float*)(base + 78708736);          // B*FC1
    float* e2    = (float*)(base + 78741504);          // B*OMIC

    const int threads = 256;
    const dim3 gemmGrid(HC / 64, NT / 128);            // 4 x 256 blocks, 8 waves each
    const int edgeBlocks  = (EP + threads - 1) / threads;
    const int edgeWBlocks = (EP * 32 + threads - 1) / threads;

    // ---- x0 = LN(mean(x)) ----
    mean_rows_kernel<<<NT / threads, threads, 0, stream>>>(x, nmean);
    ln_rows_kernel<<<Bv, threads, 0, stream>>>(nmean, ln_w, ln_b, feat, 3 * Nv, 0);

    // ================= GAT layer 1 =================
    wmma_gemm_kernel<IN_CH><<<gemmGrid, threads, 0, stream>>>(x, w1, bufA, HC);
    attn_scores_kernel<<<(NT * Hh) / threads, threads, 0, stream>>>(bufA, att_src1, att_dst1, asrc, adst);
    fill_neginf_kernel<<<(NT * Hh) / threads, threads, 0, stream>>>(amax, NT * Hh);
    hipMemsetAsync(denom, 0, (size_t)NT * Hh * 4, stream);
    hipMemsetAsync(bufB, 0, (size_t)NT * HC * 4, stream);
    edge_alpha_max_kernel<<<edgeBlocks, threads, 0, stream>>>(ei, asrc, adst, alpha, amax);
    edge_exp_sum_kernel<<<edgeBlocks, threads, 0, stream>>>(ei, alpha, amax, denom);
    edge_message_kernel<<<edgeWBlocks, threads, 0, stream>>>(ei, bufA, alpha, denom, bufB);
    elu_pool_kernel<<<NT / threads, threads, 0, stream>>>(bufB, bias1, pool1_w, pool1_b, nmean);
    ln_rows_kernel<<<Bv, threads, 0, stream>>>(nmean, ln_w, ln_b, feat, 3 * Nv, Nv);

    // ================= GAT layer 2 =================
    wmma_gemm_kernel<HC><<<gemmGrid, threads, 0, stream>>>(bufB, w2, bufA, HC);
    attn_scores_kernel<<<(NT * Hh) / threads, threads, 0, stream>>>(bufA, att_src2, att_dst2, asrc, adst);
    fill_neginf_kernel<<<(NT * Hh) / threads, threads, 0, stream>>>(amax, NT * Hh);
    hipMemsetAsync(denom, 0, (size_t)NT * Hh * 4, stream);
    hipMemsetAsync(bufB, 0, (size_t)NT * HC * 4, stream);   // h1 dead after xw2/scores
    edge_alpha_max_kernel<<<edgeBlocks, threads, 0, stream>>>(ei, asrc, adst, alpha, amax);
    edge_exp_sum_kernel<<<edgeBlocks, threads, 0, stream>>>(ei, alpha, amax, denom);
    edge_message_kernel<<<edgeWBlocks, threads, 0, stream>>>(ei, bufA, alpha, denom, bufB);
    elu_pool_kernel<<<NT / threads, threads, 0, stream>>>(bufB, bias2, pool2_w, pool2_b, nmean);
    ln_rows_kernel<<<Bv, threads, 0, stream>>>(nmean, ln_w, ln_b, feat, 3 * Nv, 2 * Nv);

    // ================= encoder =================
    fc_kernel<<<(Bv * FC0) / threads, threads, 0, stream>>>(feat, enc0_w, enc0_b, e0, Bv, 3 * Nv, FC0, 1);
    fc_kernel<<<(Bv * FC1) / threads, threads, 0, stream>>>(e0, enc1_w, enc1_b, e1, Bv, FC0, FC1, 1);
    fc_kernel<<<(Bv * OMIC) / threads, threads, 0, stream>>>(e1, enc2_w, enc2_b, e2, Bv, FC1, OMIC, 1);
    fc_kernel<<<1, threads, 0, stream>>>(e2, last_w, last_b, out, Bv, OMIC, OUTN, 0);
    (void)in_sizes; (void)n_in; (void)out_size; (void)ws_size;
}
